// Model_88184268521652
// MI455X (gfx1250) — matrix operand order repaired and checked by reading, not compiled, not run on hardware
//
#include <hip/hip_runtime.h>

typedef _Float16 f16;
typedef f16   v8h  __attribute__((ext_vector_type(8)));
typedef f16   v16h __attribute__((ext_vector_type(16)));
typedef float v8f  __attribute__((ext_vector_type(8)));

#define NTOT 16384
#define NH   8192
#define KIN  32     // 28 padded
#define HPAD 320    // 300 padded
#define OPAD 128    // 100 padded

#define WMMA_F16(a, b, c) \
  __builtin_amdgcn_wmma_f32_16x16x32_f16(false, (a), false, (b), (short)0, (c), false, false)

// ---------------- WMMA fragment loaders (wave32, 16x16x32 f16) ----------------
// A: 16x32 (MxK). Lane l<16: row=l, K 0..7 (v0-3) + 16..23 (v4-7); l>=16: row=l-16, K 8..15 + 24..31.
__device__ __forceinline__ v16h frag_a(const f16* __restrict__ base, int ld) {
  const int lane = threadIdx.x & 31;
  const f16* p = base + (lane & 15) * ld + ((lane >> 4) << 3);
  v8h lo = *(const v8h*)(p);
  v8h hi = *(const v8h*)(p + 16);
  v16h r;
#pragma unroll
  for (int i = 0; i < 8; ++i) { r[i] = lo[i]; r[8 + i] = hi[i]; }
  return r;
}

// B: 32x16 (KxN) sourced from a row-major N x K matrix (i.e. computes A * W^T).
// Lane l<16: col=l, K 0..7 + 16..23; l>=16: col=l-16, K 8..15 + 24..31.
__device__ __forceinline__ v16h frag_bt(const f16* __restrict__ base, int ld) {
  const int lane = threadIdx.x & 31;
  const f16* p = base + (lane & 15) * ld + ((lane >> 4) << 3);
  v8h lo = *(const v8h*)(p);
  v8h hi = *(const v8h*)(p + 16);
  v16h r;
#pragma unroll
  for (int i = 0; i < 8; ++i) { r[i] = lo[i]; r[8 + i] = hi[i]; }
  return r;
}

// ---------------- conversion / padding ----------------
__global__ void cvt_pad_f16(const float* __restrict__ src, f16* __restrict__ dst,
                            int rows, int cols, int pcols, int n) {
  int idx = blockIdx.x * blockDim.x + threadIdx.x;
  if (idx >= n) return;
  int r = idx / pcols, c = idx - r * pcols;
  dst[idx] = (c < cols && r < rows) ? (f16)src[r * cols + c] : (f16)0.0f;
}

__global__ void pad_bias_f32(const float* __restrict__ src, float* __restrict__ dst, int n, int np) {
  int i = blockIdx.x * blockDim.x + threadIdx.x;
  if (i < np) dst[i] = (i < n) ? src[i] : 0.0f;
}

__global__ void zero_f32(float* __restrict__ p, int n) {
  int i = blockIdx.x * blockDim.x + threadIdx.x;
  if (i < n) p[i] = 0.0f;
}

// ------- MLP layer: out = act(A @ W^T + b); one wave per 16x64 output block (1x4 tiles) -------
__global__ __launch_bounds__(256)
void gemm_bias_act(const f16* __restrict__ A, const f16* __restrict__ W,
                   const float* __restrict__ bias, f16* __restrict__ out,
                   int M, int K, int Npad, int relu) {
  int wave = (blockIdx.x * blockDim.x + threadIdx.x) >> 5;
  int ngroups = Npad >> 6;                   // groups of 4 n-tiles (Npad % 64 == 0)
  int total = (M >> 4) * ngroups;
  if (wave >= total) return;                 // whole-wave exit: EXEC stays all-ones for WMMA
  int mt = wave / ngroups, ng = wave - mt * ngroups;
  const f16* Ab = A + (size_t)(mt * 16) * K;
  const f16* Wb = W + (size_t)(ng * 64) * K;
  v8f acc0 = {}, acc1 = {}, acc2 = {}, acc3 = {};
  for (int k = 0; k < K; k += 32) {
    v16h a  = frag_a(Ab + k, K);
    v16h b0 = frag_bt(Wb + k, K);
    v16h b1 = frag_bt(Wb + (size_t)16 * K + k, K);
    v16h b2 = frag_bt(Wb + (size_t)32 * K + k, K);
    v16h b3 = frag_bt(Wb + (size_t)48 * K + k, K);
    acc0 = WMMA_F16(a, b0, acc0);
    acc1 = WMMA_F16(a, b1, acc1);
    acc2 = WMMA_F16(a, b2, acc2);
    acc3 = WMMA_F16(a, b3, acc3);
  }
  int lane = threadIdx.x & 31;
  int nl = lane & 15;
  int mbase = mt * 16 + ((lane >> 4) << 3);  // VGPR r -> row mbase + r
  v8f accs[4] = {acc0, acc1, acc2, acc3};
#pragma unroll
  for (int t = 0; t < 4; ++t) {
    int col = ng * 64 + t * 16 + nl;
    float bv = bias[col];
#pragma unroll
    for (int r = 0; r < 8; ++r) {
      float v = accs[t][r] + bv;
      if (relu) v = fmaxf(v, 0.0f);
      out[(size_t)(mbase + r) * Npad + col] = (f16)v;
    }
  }
}

// ---------------- row squared norms of f16 features (f32 accum) ----------------
__global__ void row_sqnorm(const f16* __restrict__ F, float* __restrict__ sq, int rows) {
  int i = blockIdx.x * blockDim.x + threadIdx.x;
  if (i >= rows) return;
  const f16* p = F + (size_t)i * OPAD;
  float s = 0.0f;
#pragma unroll
  for (int k = 0; k < OPAD; k += 8) {
    v8h v = *(const v8h*)(p + k);
#pragma unroll
    for (int j = 0; j < 8; ++j) { float x = (float)v[j]; s += x * x; }
  }
  sq[i] = s;
}

// ---- Gram tiles: K(i,j)=exp(-d2/sigma0^2) reduced to row/col sums + traces, never stored ----
// Each wave computes a 32x32 output block (2x2 WMMA tiles) for fragment reuse.
// grid: (jblk/8, iblk, z) with z: 0=Kx(X,X) 1=Ky(Y,Y) 2=Kxy(X,Y); 8 waves/block.
__device__ __forceinline__ void gram_epilogue(v8f acc, int rbase, int col, bool diag, int z,
                                              const float* __restrict__ sqR,
                                              const float* __restrict__ sqC,
                                              float inv, float* __restrict__ rs,
                                              float* __restrict__ csKxy,
                                              float* __restrict__ traces) {
  const int lane = threadIdx.x & 31;
  const int nl = lane & 15;
  const int off = (lane >> 4) << 3;
  const float sqc = sqC[col];
  float vals[8];
  float colsum = 0.0f;
#pragma unroll
  for (int r = 0; r < 8; ++r) {
    float d = sqR[rbase + r] + sqc - 2.0f * acc[r];
    d = fmaxf(d, 0.0f);
    float kv = __expf(d * inv);
    vals[r] = kv;
    colsum += kv;
  }
  // row sums: reduce across the 16-lane half-group holding this row
#pragma unroll
  for (int r = 0; r < 8; ++r) {
    float v = vals[r];
    v += __shfl_xor(v, 1, 32);
    v += __shfl_xor(v, 2, 32);
    v += __shfl_xor(v, 4, 32);
    v += __shfl_xor(v, 8, 32);
    if (nl == 0) unsafeAtomicAdd(&rs[rbase + r], v);
  }
  // column sums (only Kxy: hh needs Kxy^T row sums = Kxy col sums)
  if (z == 2) {
    float c = colsum + __shfl_xor(colsum, 16, 32);
    if (lane < 16) unsafeAtomicAdd(&csKxy[col], c);
  }
  // traces from diagonal tiles
  if (diag) {
    float t = 0.0f;
#pragma unroll
    for (int r = 0; r < 8; ++r)
      if (r + off == nl) t = vals[r];
    t += __shfl_xor(t, 16, 32);
    t += __shfl_xor(t, 8, 32);
    t += __shfl_xor(t, 4, 32);
    t += __shfl_xor(t, 2, 32);
    t += __shfl_xor(t, 1, 32);
    if (lane == 0) unsafeAtomicAdd(&traces[z], t);
  }
}

__global__ __launch_bounds__(256)
void gram_kernel(const f16* __restrict__ F, const float* __restrict__ sq,
                 const float* __restrict__ sigma0OPT,
                 float* __restrict__ rsKx, float* __restrict__ rsKy,
                 float* __restrict__ rsKxy, float* __restrict__ csKxy,
                 float* __restrict__ traces) {
  const int z = blockIdx.z;
  const int wave = threadIdx.x >> 5;
  const int jb = blockIdx.x * 8 + wave;     // 32-wide column block (0..255)
  const int ib = blockIdx.y;                // 32-tall row block (0..255)

  const f16* Rp = F + (size_t)(z == 1 ? NH * OPAD : 0);   // rows side (X, Y, X)
  const f16* Cp = F + (size_t)(z == 0 ? 0 : NH * OPAD);   // cols side (X, Y, Y)
  const float* sqR = sq + (z == 1 ? NH : 0);
  const float* sqC = sq + (z == 0 ? 0 : NH);
  float* rs = (z == 0) ? rsKx : ((z == 1) ? rsKy : rsKxy);

  const f16* Ab0 = Rp + (size_t)(ib * 32) * OPAD;
  const f16* Ab1 = Ab0 + (size_t)16 * OPAD;
  const f16* Bb0 = Cp + (size_t)(jb * 32) * OPAD;
  const f16* Bb1 = Bb0 + (size_t)16 * OPAD;

  v8f acc00 = {}, acc01 = {}, acc10 = {}, acc11 = {};
#pragma unroll
  for (int k = 0; k < OPAD; k += 32) {
    v16h a0 = frag_a(Ab0 + k, OPAD);
    v16h a1 = frag_a(Ab1 + k, OPAD);
    v16h b0 = frag_bt(Bb0 + k, OPAD);
    v16h b1 = frag_bt(Bb1 + k, OPAD);
    acc00 = WMMA_F16(a0, b0, acc00);
    acc01 = WMMA_F16(a0, b1, acc01);
    acc10 = WMMA_F16(a1, b0, acc10);
    acc11 = WMMA_F16(a1, b1, acc11);
  }

  const int lane = threadIdx.x & 31;
  const int nl = lane & 15;
  const int off = (lane >> 4) << 3;
  const float s0 = sigma0OPT[0];
  const float inv = -1.0f / (s0 * s0);

  const int it0 = ib * 2, jt0 = jb * 2;     // 16-granular tile indices
  gram_epilogue(acc00, it0 * 16 + off,        jt0 * 16 + nl,        it0 == jt0,         z, sqR, sqC, inv, rs, csKxy, traces);
  gram_epilogue(acc01, it0 * 16 + off,        (jt0 + 1) * 16 + nl,  it0 == jt0 + 1,     z, sqR, sqC, inv, rs, csKxy, traces);
  gram_epilogue(acc10, (it0 + 1) * 16 + off,  jt0 * 16 + nl,        it0 + 1 == jt0,     z, sqR, sqC, inv, rs, csKxy, traces);
  gram_epilogue(acc11, (it0 + 1) * 16 + off,  (jt0 + 1) * 16 + nl,  it0 == jt0,         z, sqR, sqC, inv, rs, csKxy, traces);
}

// ---------------- finalize: MMD statistic ----------------
__global__ __launch_bounds__(256)
void finalize_kernel(const float* __restrict__ rsKx, const float* __restrict__ rsKy,
                     const float* __restrict__ rsKxy, const float* __restrict__ csKxy,
                     const float* __restrict__ traces, float* __restrict__ out) {
  __shared__ float sh[256];
  const float nf = (float)NH;
  float sx = 0.f, sy = 0.f, sxy = 0.f, v1 = 0.f;
  for (int i = threadIdx.x; i < NH; i += 256) {
    float a = rsKx[i], b = rsKy[i], c = rsKxy[i], d = csKxy[i];
    sx += a; sy += b; sxy += c;
    float row = (a + b - c - d) / nf;
    v1 += row * row;
  }
  float vals[4] = {sx, sy, sxy, v1};
  float red[4];
#pragma unroll
  for (int q = 0; q < 4; ++q) {
    sh[threadIdx.x] = vals[q];
    __syncthreads();
    for (int s = 128; s > 0; s >>= 1) {
      if (threadIdx.x < s) sh[threadIdx.x] += sh[threadIdx.x + s];
      __syncthreads();
    }
    red[q] = sh[0];
    __syncthreads();
  }
  if (threadIdx.x == 0) {
    float SKx = red[0], SKy = red[1], SKxy = red[2], V1 = red[3] / nf;
    float denom = nf * (nf - 1.0f);
    float xx = (SKx - traces[0]) / denom;
    float yy = (SKy - traces[1]) / denom;
    float xy = (SKxy - traces[2]) / denom;
    float mmd2 = xx - 2.0f * xy + yy;
    float V2 = (SKx + SKy - 2.0f * SKxy) / (nf * nf);
    float varEst = 4.0f * (V1 - V2 * V2);
    out[0] = -(mmd2 / sqrtf(varEst + 1e-8f));
  }
}

// ---------------- host orchestration ----------------
extern "C" void kernel_launch(void* const* d_in, const int* in_sizes, int n_in,
                              void* d_out, int out_size, void* d_ws, size_t ws_size,
                              hipStream_t stream) {
  const float* XY = (const float*)d_in[0];
  const float* w[6] = {(const float*)d_in[1], (const float*)d_in[3], (const float*)d_in[5],
                       (const float*)d_in[7], (const float*)d_in[9], (const float*)d_in[11]};
  const float* b[6] = {(const float*)d_in[2], (const float*)d_in[4], (const float*)d_in[6],
                       (const float*)d_in[8], (const float*)d_in[10], (const float*)d_in[12]};
  const float* sigma0 = (const float*)d_in[13];

  char* ws = (char*)d_ws;
  size_t off = 0;
  auto take = [&](size_t bytes) -> void* {
    void* p = ws + off;
    off += (bytes + 255) & ~(size_t)255;
    return p;
  };

  f16* in16 = (f16*)take((size_t)NTOT * KIN * sizeof(f16));
  f16* act0 = (f16*)take((size_t)NTOT * HPAD * sizeof(f16));
  f16* act1 = (f16*)take((size_t)NTOT * HPAD * sizeof(f16));
  f16* fbuf = (f16*)take((size_t)NTOT * OPAD * sizeof(f16));
  f16* wh[6];
  wh[0] = (f16*)take((size_t)HPAD * KIN * sizeof(f16));
  for (int i = 1; i < 5; ++i) wh[i] = (f16*)take((size_t)HPAD * HPAD * sizeof(f16));
  wh[5] = (f16*)take((size_t)OPAD * HPAD * sizeof(f16));
  float* bp[6];
  for (int i = 0; i < 5; ++i) bp[i] = (float*)take(HPAD * sizeof(float));
  bp[5] = (float*)take(OPAD * sizeof(float));
  float* sqn = (float*)take((size_t)NTOT * sizeof(float));
  float* accum = (float*)take((size_t)(4 * NH + 16) * sizeof(float));
  float* rsKx = accum;
  float* rsKy = accum + NH;
  float* rsKxy = accum + 2 * NH;
  float* csKxy = accum + 3 * NH;
  float* traces = accum + 4 * NH;

  // zero accumulators (workspace is not re-zeroed between replays)
  {
    int n = 4 * NH + 16;
    zero_f32<<<(n + 255) / 256, 256, 0, stream>>>(accum, n);
  }

  // convert + pad input / weights / biases
  auto cvt = [&](const float* src, f16* dst, int rows, int cols, int prows, int pcols) {
    int n = prows * pcols;
    cvt_pad_f16<<<(n + 255) / 256, 256, 0, stream>>>(src, dst, rows, cols, pcols, n);
  };
  cvt(XY, in16, NTOT, 28, NTOT, KIN);
  cvt(w[0], wh[0], 300, 28, HPAD, KIN);
  for (int i = 1; i < 5; ++i) cvt(w[i], wh[i], 300, 300, HPAD, HPAD);
  cvt(w[5], wh[5], 100, 300, OPAD, HPAD);
  for (int i = 0; i < 5; ++i)
    pad_bias_f32<<<(HPAD + 255) / 256, 256, 0, stream>>>(b[i], bp[i], 300, HPAD);
  pad_bias_f32<<<(OPAD + 255) / 256, 256, 0, stream>>>(b[5], bp[5], 100, OPAD);

  // MLP (each wave: 16 rows x 64 cols)
  auto gemm = [&](const f16* A, const f16* W, const float* bias, f16* out,
                  int M, int K, int Npad, int relu) {
    int waves = (M >> 4) * (Npad >> 6);
    int blocks = (waves * 32 + 255) / 256;
    gemm_bias_act<<<blocks, 256, 0, stream>>>(A, W, bias, out, M, K, Npad, relu);
  };
  gemm(in16, wh[0], bp[0], act0, NTOT, KIN, HPAD, 1);
  gemm(act0, wh[1], bp[1], act1, NTOT, HPAD, HPAD, 1);
  gemm(act1, wh[2], bp[2], act0, NTOT, HPAD, HPAD, 1);
  gemm(act0, wh[3], bp[3], act1, NTOT, HPAD, HPAD, 1);
  gemm(act1, wh[4], bp[4], act0, NTOT, HPAD, HPAD, 1);
  gemm(act0, wh[5], bp[5], fbuf, NTOT, HPAD, OPAD, 0);

  // squared norms
  row_sqnorm<<<(NTOT + 255) / 256, 256, 0, stream>>>(fbuf, sqn, NTOT);

  // Gram reductions: 256x256 blocks of 32x32 (2x2 WMMA tiles per wave), z = {Kx,Ky,Kxy}
  gram_kernel<<<dim3(NH / 32 / 8, NH / 32, 3), 256, 0, stream>>>(
      fbuf, sqn, sigma0, rsKx, rsKy, rsKxy, csKxy, traces);

  // finalize
  finalize_kernel<<<1, 256, 0, stream>>>(rsKx, rsKy, rsKxy, csKxy, traces, (float*)d_out);
}
